// AxialAttentionBlock_2911987827009
// MI455X (gfx1250) — matrix operand order repaired and checked by reading, not compiled, not run on hardware
//
#include <hip/hip_runtime.h>
#include <hip/hip_bf16.h>
#include <math.h>

#define cB 2
#define cC 768
#define cD 8
#define cH 64
#define cW 64
#define cS (cD * cH * cW)   /* 32768 */
#define cHE 12
#define cHD 64

typedef __attribute__((ext_vector_type(16))) __bf16 bf16x16;
typedef __attribute__((ext_vector_type(8)))  float  v8f;
typedef __attribute__((ext_vector_type(4)))  unsigned int u32x4;

static constexpr size_t BCS = (size_t)cB * cC * cS;  // 50,331,648

__device__ __forceinline__ bf16x16 frag_ld(const __bf16* p0, const __bf16* p1) {
  union { u32x4 u[2]; bf16x16 v; } f;
  f.u[0] = *reinterpret_cast<const u32x4*>(p0);
  f.u[1] = *reinterpret_cast<const u32x4*>(p1);
  return f.v;
}

__device__ __forceinline__ v8f wmma_bf16(bf16x16 a, bf16x16 b, v8f c) {
  return __builtin_amdgcn_wmma_f32_16x16x32_bf16(false, a, false, b, (short)0, c,
                                                 false, false);
}

// CDNA5 async global->LDS copy (GV form, 16B per lane). The low 32 bits of a
// generic pointer to __shared__ are the group-segment offset (flat aperture
// truncates addr[31:0]), which is exactly what the VDST LDS-address VGPR wants.
__device__ __forceinline__ void async_ld_b128(void* lds, const void* gaddr) {
  unsigned lds_off = (unsigned)(size_t)lds;
  unsigned long long ga = (unsigned long long)(size_t)gaddr;
  asm volatile("global_load_async_to_lds_b128 %0, %1, off"
               :: "v"(lds_off), "v"(ga)
               : "memory");
}
__device__ __forceinline__ void wait_async0() {
  asm volatile("s_wait_asynccnt 0" ::: "memory");
}

// ---------------------------------------------------------------------------
// RMS-instance-norm statistics: inv[bc] = rsqrt(sum(x^2)*mul + eps)
// mul folds both 1/S and any prescale^2 (e.g. the 1/9 from the /3 average).
// ---------------------------------------------------------------------------
__global__ __launch_bounds__(256) void k_rms_stats(const float* __restrict__ x,
                                                   float* __restrict__ inv,
                                                   float mul, float eps) {
  const int bc = blockIdx.x;
  const float* p = x + (size_t)bc * cS;
  float s = 0.f;
  for (int i = threadIdx.x; i < cS; i += 256) { float t = p[i]; s += t * t; }
  __shared__ float red[256];
  red[threadIdx.x] = s;
  __syncthreads();
  for (int w = 128; w > 0; w >>= 1) {
    if ((int)threadIdx.x < w) red[threadIdx.x] += red[threadIdx.x + w];
    __syncthreads();
  }
  if (threadIdx.x == 0) inv[bc] = rsqrtf(red[0] * mul + eps);
}

// out = bf16(x * prescale * inv[bc] * w[c])
__global__ __launch_bounds__(256) void k_norm_cast(const float* __restrict__ x,
                                                   const float* __restrict__ inv,
                                                   const float* __restrict__ w,
                                                   __bf16* __restrict__ out,
                                                   float prescale) {
  const size_t n = BCS;
  for (size_t i = (size_t)blockIdx.x * 256 + threadIdx.x; i < n;
       i += (size_t)gridDim.x * 256) {
    size_t bc = i / cS;
    int c = (int)(bc % cC);
    out[i] = (__bf16)(x[i] * prescale * inv[bc] * w[c]);
  }
}

__global__ __launch_bounds__(256) void k_cast_bf16(const float* __restrict__ in,
                                                   __bf16* __restrict__ out,
                                                   size_t n) {
  for (size_t i = (size_t)blockIdx.x * 256 + threadIdx.x; i < n;
       i += (size_t)gridDim.x * 256)
    out[i] = (__bf16)in[i];
}

// ---------------------------------------------------------------------------
// bf16 WMMA GEMM:  Out[b][M][N] = W[M,K] x X[b][K,N]  (+ epilogue)
// Block: 256 threads = 8 waves, block tile 128x256, wave tile 64x64,
// 16 x v_wmma_f32_16x16x32_bf16 per wave per K-step of 32.
// A tile is DMAed with global_load_async_to_lds_b128 (ASYNCcnt).
// EPI 0: +bias -> f32 ; 1: res + gamma*( +bias) -> f32 ;
// EPI 2: gelu(+bias) -> bf16 ; 3: +bias -> f32.
// ---------------------------------------------------------------------------
template <int EPI>
__global__ __launch_bounds__(256) void k_gemm(const __bf16* __restrict__ Wm,
                                              const __bf16* __restrict__ Xm,
                                              void* __restrict__ Outv,
                                              const float* __restrict__ bias,
                                              const float* __restrict__ gamma,
                                              const float* __restrict__ res,
                                              int M, int N, int K) {
  const int bb = blockIdx.z;
  const __bf16* X = Xm + (size_t)bb * K * N;
  const int m0 = blockIdx.y * 128;
  const int n0 = blockIdx.x * 256;
  const int tid = threadIdx.x;
  const int lane = tid & 31, wv = tid >> 5;
  const int wm = wv >> 2, wn = wv & 3;  // 2 x 4 wave grid
  const int l16 = lane & 15, lhi = lane >> 4;

  __shared__ __align__(16) __bf16 sA[128][40];  // [M][K] K-contig, pad 8
  __shared__ __align__(16) __bf16 sB[256][40];  // X tile transposed: [N][K]

  v8f acc[4][4];
#pragma unroll
  for (int i = 0; i < 4; ++i)
#pragma unroll
    for (int j = 0; j < 4; ++j)
#pragma unroll
      for (int r = 0; r < 8; ++r) acc[i][j][r] = 0.f;

  for (int k0 = 0; k0 < K; k0 += 32) {
    // A tile 128x32: 2 chunks of 8 bf16 per thread, async DMA into LDS
#pragma unroll
    for (int c = 0; c < 2; ++c) {
      int id = tid * 2 + c;
      int row = id >> 2, kc = (id & 3) * 8;
      async_ld_b128(&sA[row][kc], &Wm[(size_t)(m0 + row) * K + k0 + kc]);
    }
    // X tile 32x256, stored transposed -> sB[n][k]
    {
      int krow = tid >> 3;          // 0..31
      int nc0 = (tid & 7) * 32;     // 0..224
#pragma unroll
      for (int c = 0; c < 4; ++c) {
        u32x4 d = *reinterpret_cast<const u32x4*>(
            &X[(size_t)(k0 + krow) * N + n0 + nc0 + c * 8]);
        const __bf16* e = reinterpret_cast<const __bf16*>(&d);
#pragma unroll
        for (int t = 0; t < 8; ++t) sB[nc0 + c * 8 + t][krow] = e[t];
      }
    }
    if (k0 + 32 < K) {  // emits global_prefetch_b8 for the next tiles
      __builtin_prefetch(&Wm[(size_t)(m0 + (tid >> 1)) * K + k0 + 32], 0, 0);
      __builtin_prefetch(&X[(size_t)(k0 + 32 + (tid >> 3)) * N + n0 + (tid & 7) * 32], 0, 0);
    }
    wait_async0();
    __syncthreads();

    bf16x16 af[4], bfr[4];
#pragma unroll
    for (int i = 0; i < 4; ++i) {
      // A frag: lanes 0-15 hold K {0..7,16..23}, lanes 16-31 K {8..15,24..31}
      const __bf16* rp = &sA[wm * 64 + i * 16 + l16][lhi ? 8 : 0];
      af[i] = frag_ld(rp, rp + 16);
    }
#pragma unroll
    for (int j = 0; j < 4; ++j) {
      // B frag: lane n holds column n, K {0..7,16..23} / {8..15,24..31}
      const __bf16* rp = &sB[wn * 64 + j * 16 + l16][lhi ? 8 : 0];
      bfr[j] = frag_ld(rp, rp + 16);
    }
#pragma unroll
    for (int i = 0; i < 4; ++i)
#pragma unroll
      for (int j = 0; j < 4; ++j) acc[i][j] = wmma_bf16(af[i], bfr[j], acc[i][j]);
    __syncthreads();
  }

  const size_t ob = (size_t)bb * M * N;
#pragma unroll
  for (int i = 0; i < 4; ++i)
#pragma unroll
    for (int j = 0; j < 4; ++j) {
      const int col = n0 + wn * 64 + j * 16 + l16;
#pragma unroll
      for (int r = 0; r < 8; ++r) {
        const int row = m0 + wm * 64 + i * 16 + r + (lhi ? 8 : 0);
        float v = acc[i][j][r] + bias[row];
        const size_t off = ob + (size_t)row * N + col;
        if (EPI == 0) {
          ((float*)Outv)[off] = v;
        } else if (EPI == 1) {
          ((float*)Outv)[off] = res[off] + gamma[row] * v;
        } else if (EPI == 2) {
          float g = 0.5f * v * (1.f + erff(v * 0.70710678118f));
          ((__bf16*)Outv)[off] = (__bf16)g;
        } else {
          ((float*)Outv)[off] = v;
        }
      }
    }
}

// ---------------------------------------------------------------------------
// LayerNorm q,k per token/head + repack qkv f32 [b][3C][S] ->
// q/k/v bf16 [b*HE][S][HD] (token rows contiguous -> direct WMMA operands)
// ---------------------------------------------------------------------------
__global__ __launch_bounds__(256) void k_ln_pack(const float* __restrict__ qkv,
                                                 const float* __restrict__ qw,
                                                 const float* __restrict__ qb,
                                                 const float* __restrict__ kw,
                                                 const float* __restrict__ kb,
                                                 __bf16* __restrict__ Qo,
                                                 __bf16* __restrict__ Ko,
                                                 __bf16* __restrict__ Vo) {
  const size_t idx = (size_t)blockIdx.x * 256 + threadIdx.x;  // [0, B*HE*S)
  const int s = (int)(idx % cS);
  const int bh = (int)(idx / cS);
  const int he = bh % cHE;
  const int b = bh / cHE;
  const float* src = qkv + ((size_t)b * 3 * cC + (size_t)he * 3 * cHD) * cS + s;
  __bf16* qo = Qo + idx * cHD;
  __bf16* ko = Ko + idx * cHD;
  __bf16* vo = Vo + idx * cHD;
  float v[cHD];
  // --- q ---
  float sum = 0.f;
#pragma unroll
  for (int j = 0; j < cHD; ++j) { v[j] = src[(size_t)j * cS]; sum += v[j]; }
  float mu = sum * (1.f / cHD);
  float var = 0.f;
#pragma unroll
  for (int j = 0; j < cHD; ++j) { float d = v[j] - mu; var += d * d; }
  float inv = rsqrtf(var * (1.f / cHD) + 1e-5f);
#pragma unroll
  for (int j = 0; j < cHD; ++j) qo[j] = (__bf16)((v[j] - mu) * inv * qw[j] + qb[j]);
  // --- k ---
  sum = 0.f;
#pragma unroll
  for (int j = 0; j < cHD; ++j) { v[j] = src[(size_t)(cHD + j) * cS]; sum += v[j]; }
  mu = sum * (1.f / cHD);
  var = 0.f;
#pragma unroll
  for (int j = 0; j < cHD; ++j) { float d = v[j] - mu; var += d * d; }
  inv = rsqrtf(var * (1.f / cHD) + 1e-5f);
#pragma unroll
  for (int j = 0; j < cHD; ++j) ko[j] = (__bf16)((v[j] - mu) * inv * kw[j] + kb[j]);
  // --- v ---
#pragma unroll
  for (int j = 0; j < cHD; ++j) vo[j] = (__bf16)src[(size_t)(2 * cHD + j) * cS];
}

// ---------------------------------------------------------------------------
// Axial attention for a 64-token sequence (W axis: stride 1; H axis: stride W).
// One workgroup (8 waves) per (b,he,seq). Whole Q/K/V tiles in LDS (Q/K via
// async DMA); QK^T and PV are each 16 WMMA tiles (2 per wave). Output layout:
// o[b][c][s] so the next RMS norm + out-proj GEMM read it channel-major.
// mode 0=store, 1=add.
// ---------------------------------------------------------------------------
__global__ __launch_bounds__(256) void k_attn64(const __bf16* __restrict__ Qb,
                                                const __bf16* __restrict__ Kb,
                                                const __bf16* __restrict__ Vb,
                                                float* __restrict__ O,
                                                int axis, int mode) {
  const int blk = blockIdx.x;
  const int seq = blk % (cD * 64);
  const int bh = blk / (cD * 64);
  const int he = bh % cHE;
  const int b = bh / cHE;
  int s0, stride;
  if (axis == 0) { s0 = seq * cW; stride = 1; }            // seq = d*H + h
  else { int d = seq / cW, w = seq % cW; s0 = d * cH * cW + w; stride = cW; }
  const size_t base = (size_t)bh * cS * cHD;
  const size_t obase = ((size_t)b * cC + (size_t)he * cHD) * cS;
  const int tid = threadIdx.x, lane = tid & 31, wv = tid >> 5;
  const int l16 = lane & 15, lhi = lane >> 4;

  __shared__ __align__(16) __bf16 sQ[64][72];   // Q, then reused for P
  __shared__ __align__(16) __bf16 sK[64][72];
  __shared__ __align__(16) __bf16 sVt[64][72];  // V transposed [hd][krow]
  __shared__ float sL[64][64];

  {
    const int row = tid >> 2, cc = (tid & 3) * 16;
    const size_t g = base + (size_t)(s0 + row * stride) * cHD + cc;
    async_ld_b128(&sQ[row][cc],     &Qb[g]);
    async_ld_b128(&sQ[row][cc + 8], &Qb[g + 8]);
    async_ld_b128(&sK[row][cc],     &Kb[g]);
    async_ld_b128(&sK[row][cc + 8], &Kb[g + 8]);
    union { u32x4 u[2]; __bf16 e[16]; } vvu;
    vvu.u[0] = *reinterpret_cast<const u32x4*>(&Vb[g]);
    vvu.u[1] = *reinterpret_cast<const u32x4*>(&Vb[g + 8]);
#pragma unroll
    for (int t = 0; t < 16; ++t) sVt[cc + t][row] = vvu.e[t];
  }
  wait_async0();
  __syncthreads();

  // logits = (Q K^T) * HD^-0.5
#pragma unroll
  for (int pass = 0; pass < 2; ++pass) {
    int tt = wv + pass * 8;
    int qi = tt >> 2, kj = tt & 3;
    v8f a8;
#pragma unroll
    for (int r = 0; r < 8; ++r) a8[r] = 0.f;
#pragma unroll
    for (int kk = 0; kk < 64; kk += 32) {
      const __bf16* qp = &sQ[qi * 16 + l16][kk + (lhi ? 8 : 0)];
      bf16x16 a = frag_ld(qp, qp + 16);
      const __bf16* kp = &sK[kj * 16 + l16][kk + (lhi ? 8 : 0)];
      bf16x16 bv = frag_ld(kp, kp + 16);
      a8 = wmma_bf16(a, bv, a8);
    }
#pragma unroll
    for (int r = 0; r < 8; ++r)
      sL[qi * 16 + r + (lhi ? 8 : 0)][kj * 16 + l16] = a8[r] * 0.125f;
  }
  __syncthreads();

  // row softmax -> P (bf16, reuse sQ)
  if (tid < 64) {
    float m = -3.4e38f;
#pragma unroll
    for (int k = 0; k < 64; ++k) m = fmaxf(m, sL[tid][k]);
    float e[64], sum = 0.f;
#pragma unroll
    for (int k = 0; k < 64; ++k) { e[k] = __expf(sL[tid][k] - m); sum += e[k]; }
    float is = 1.f / sum;
#pragma unroll
    for (int k = 0; k < 64; ++k) sQ[tid][k] = (__bf16)(e[k] * is);
  }
  __syncthreads();

  // O = P @ V
#pragma unroll
  for (int pass = 0; pass < 2; ++pass) {
    int tt = wv + pass * 8;
    int qi = tt >> 2, hj = tt & 3;
    v8f a8;
#pragma unroll
    for (int r = 0; r < 8; ++r) a8[r] = 0.f;
#pragma unroll
    for (int kk = 0; kk < 64; kk += 32) {
      const __bf16* pp = &sQ[qi * 16 + l16][kk + (lhi ? 8 : 0)];
      bf16x16 a = frag_ld(pp, pp + 16);
      const __bf16* vp = &sVt[hj * 16 + l16][kk + (lhi ? 8 : 0)];
      bf16x16 bv = frag_ld(vp, vp + 16);
      a8 = wmma_bf16(a, bv, a8);
    }
#pragma unroll
    for (int r = 0; r < 8; ++r) {
      int qrow = qi * 16 + r + (lhi ? 8 : 0);
      int hd = hj * 16 + l16;
      size_t off = obase + (size_t)hd * cS + (size_t)(s0 + qrow * stride);
      if (mode == 0) O[off] = a8[r];
      else O[off] += a8[r];
    }
  }
}

// D-axis attention (L=8): ~0.2% of the FLOPs -> one thread per sequence.
__global__ __launch_bounds__(256) void k_attn_d(const __bf16* __restrict__ Qb,
                                                const __bf16* __restrict__ Kb,
                                                const __bf16* __restrict__ Vb,
                                                float* __restrict__ O) {
  const int idx = blockIdx.x * 256 + threadIdx.x;  // [0, B*HE*H*W)
  const int hw = idx % (cH * cW);
  const int bh = idx / (cH * cW);
  const int he = bh % cHE;
  const int b = bh / cHE;
  const size_t base = (size_t)bh * cS * cHD;
  const size_t obase = ((size_t)b * cC + (size_t)he * cHD) * cS;
  float l[8][8];
#pragma unroll
  for (int i = 0; i < 8; ++i)
#pragma unroll
    for (int j = 0; j < 8; ++j) l[i][j] = 0.f;
  for (int hd = 0; hd < cHD; ++hd) {
    float qv[8], kv[8];
#pragma unroll
    for (int d = 0; d < 8; ++d) {
      size_t off = base + (size_t)(d * cH * cW + hw) * cHD + hd;
      qv[d] = (float)Qb[off];
      kv[d] = (float)Kb[off];
    }
#pragma unroll
    for (int i = 0; i < 8; ++i)
#pragma unroll
      for (int j = 0; j < 8; ++j) l[i][j] += qv[i] * kv[j];
  }
  float p[8][8];
#pragma unroll
  for (int i = 0; i < 8; ++i) {
    float m = -3.4e38f;
#pragma unroll
    for (int j = 0; j < 8; ++j) { l[i][j] *= 0.125f; m = fmaxf(m, l[i][j]); }
    float sum = 0.f;
#pragma unroll
    for (int j = 0; j < 8; ++j) { p[i][j] = __expf(l[i][j] - m); sum += p[i][j]; }
    float is = 1.f / sum;
#pragma unroll
    for (int j = 0; j < 8; ++j) p[i][j] *= is;
  }
  for (int hd = 0; hd < cHD; ++hd) {
    float vv[8];
#pragma unroll
    for (int d = 0; d < 8; ++d)
      vv[d] = (float)Vb[base + (size_t)(d * cH * cW + hw) * cHD + hd];
#pragma unroll
    for (int i = 0; i < 8; ++i) {
      float a = 0.f;
#pragma unroll
      for (int j = 0; j < 8; ++j) a += p[i][j] * vv[j];
      O[obase + (size_t)hd * cS + (size_t)(i * cH * cW + hw)] += a;
    }
  }
}

// out = x2 + gamma_mlp[c] * (hm * inv3[bc] * mlp_norm_w[c])
__global__ __launch_bounds__(256) void k_final(const float* __restrict__ x2,
                                               const float* __restrict__ hm,
                                               const float* __restrict__ inv3,
                                               const float* __restrict__ mlpw,
                                               const float* __restrict__ gamma,
                                               float* __restrict__ out) {
  const size_t n = BCS;
  for (size_t i = (size_t)blockIdx.x * 256 + threadIdx.x; i < n;
       i += (size_t)gridDim.x * 256) {
    size_t bc = i / cS;
    int c = (int)(bc % cC);
    out[i] = x2[i] + gamma[c] * (hm[i] * inv3[bc] * mlpw[c]);
  }
}

// ---------------------------------------------------------------------------
extern "C" void kernel_launch(void* const* d_in, const int* in_sizes, int n_in,
                              void* d_out, int out_size, void* d_ws, size_t ws_size,
                              hipStream_t stream) {
  (void)in_sizes; (void)n_in; (void)out_size; (void)ws_size;
  const float* x         = (const float*)d_in[0];
  const float* norm1_w   = (const float*)d_in[1];
  const float* norm2_w   = (const float*)d_in[2];
  const float* mlp_norm_w= (const float*)d_in[3];
  const float* qkv_w     = (const float*)d_in[4];
  const float* qkv_b     = (const float*)d_in[5];
  const float* out_w     = (const float*)d_in[6];
  const float* out_b     = (const float*)d_in[7];
  const float* qnorm_w   = (const float*)d_in[8];
  const float* qnorm_b   = (const float*)d_in[9];
  const float* knorm_w   = (const float*)d_in[10];
  const float* knorm_b   = (const float*)d_in[11];
  const float* gamma_att = (const float*)d_in[12];
  const float* gamma_mlp = (const float*)d_in[13];
  const float* fc1_w     = (const float*)d_in[14];
  const float* fc1_b     = (const float*)d_in[15];
  const float* fc2_w     = (const float*)d_in[16];
  const float* fc2_b     = (const float*)d_in[17];
  // d_in[18] (bcs) unused

  char* ws = (char*)d_ws;
  size_t off = 0;
  auto bump = [&](size_t bytes) -> char* {
    char* p = ws + off;
    off += (bytes + 255) & ~(size_t)255;
    return p;
  };
  float*  inv1 = (float*)bump((size_t)cB * cC * 4);
  float*  inv2 = (float*)bump((size_t)cB * cC * 4);
  float*  inv3 = (float*)bump((size_t)cB * cC * 4);
  __bf16* wq   = (__bf16*)bump((size_t)3 * cC * cC * 2);
  __bf16* wo   = (__bf16*)bump((size_t)cC * cC * 2);
  __bf16* w1   = (__bf16*)bump((size_t)4 * cC * cC * 2);
  __bf16* w2   = (__bf16*)bump((size_t)cC * 4 * cC * 2);
  __bf16* h1   = (__bf16*)bump(BCS * 2);            // h1 -> o_n -> xc (bf16)
  char*   r3   = bump((size_t)cB * 3 * cC * cS * 4);// qkv f32 -> hmid bf16 + hm f32
  float*  qkvf = (float*)r3;
  __bf16* hmid = (__bf16*)r3;
  float*  hm   = (float*)(r3 + (size_t)cB * 4 * cC * cS * 2);
  __bf16* qbf  = (__bf16*)bump(BCS * 2);
  __bf16* kbf  = (__bf16*)bump(BCS * 2);
  __bf16* vbf  = (__bf16*)bump(BCS * 2);
  float*  r5   = (float*)bump(BCS * 4);             // o f32 -> x2 f32
  float*  o    = r5;
  float*  x2   = r5;

  // 1) norm1 + cast, weight casts
  k_rms_stats<<<cB * cC, 256, 0, stream>>>(x, inv1, 1.f / cS, 1e-6f);
  k_norm_cast<<<2048, 256, 0, stream>>>(x, inv1, norm1_w, h1, 1.f);
  k_cast_bf16<<<2048, 256, 0, stream>>>(qkv_w, wq, (size_t)3 * cC * cC);
  k_cast_bf16<<<512, 256, 0, stream>>>(out_w, wo, (size_t)cC * cC);
  k_cast_bf16<<<2048, 256, 0, stream>>>(fc1_w, w1, (size_t)4 * cC * cC);
  k_cast_bf16<<<2048, 256, 0, stream>>>(fc2_w, w2, (size_t)cC * 4 * cC);

  // 2) QKV projection (WMMA), per-head q/k LN + pack
  k_gemm<0><<<dim3(cS / 256, (3 * cC) / 128, cB), 256, 0, stream>>>(
      wq, h1, qkvf, qkv_b, nullptr, nullptr, 3 * cC, cS, cC);
  k_ln_pack<<<(cB * cHE * cS) / 256, 256, 0, stream>>>(
      qkvf, qnorm_w, qnorm_b, knorm_w, knorm_b, qbf, kbf, vbf);

  // 3) axial attention: W (store), H (add), D (add). /3 folded into norm2.
  k_attn64<<<cB * cHE * cD * 64, 256, 0, stream>>>(qbf, kbf, vbf, o, 0, 0);
  k_attn64<<<cB * cHE * cD * 64, 256, 0, stream>>>(qbf, kbf, vbf, o, 1, 1);
  k_attn_d<<<(cB * cHE * cH * cW) / 256, 256, 0, stream>>>(qbf, kbf, vbf, o);

  // 4) norm2 (with prescale 1/3) + out-proj with fused residual + gamma_att
  k_rms_stats<<<cB * cC, 256, 0, stream>>>(o, inv2, (1.f / 9.f) / cS, 1e-6f);
  k_norm_cast<<<2048, 256, 0, stream>>>(o, inv2, norm2_w, h1, 1.f / 3.f);
  k_gemm<1><<<dim3(cS / 256, cC / 128, cB), 256, 0, stream>>>(
      wo, h1, x2, out_b, gamma_att, x, cC, cS, cC);

  // 5) MLP: fc1 (+exact GELU -> bf16), fc2, mlp RMS norm, final residual
  k_cast_bf16<<<2048, 256, 0, stream>>>(x2, h1, BCS);
  k_gemm<2><<<dim3(cS / 256, (4 * cC) / 128, cB), 256, 0, stream>>>(
      w1, h1, hmid, fc1_b, nullptr, nullptr, 4 * cC, cS, cC);
  k_gemm<3><<<dim3(cS / 256, cC / 128, cB), 256, 0, stream>>>(
      w2, hmid, hm, fc2_b, nullptr, nullptr, cC, cS, 4 * cC);
  k_rms_stats<<<cB * cC, 256, 0, stream>>>(hm, inv3, 1.f / cS, 1e-6f);
  k_final<<<2048, 256, 0, stream>>>(x2, hm, inv3, mlp_norm_w, gamma_mlp,
                                    (float*)d_out);
}
